// MultiHeadAttention_58317065945795
// MI455X (gfx1250) — matrix operand order repaired and checked by reading, not compiled, not run on hardware
//
#include <hip/hip_runtime.h>
#include <stdint.h>

// ---------------------------------------------------------------------------
// MHA with RoPE for MI455X (gfx1250): bf16 WMMA everywhere, fp32 accumulate.
// H=16, DK=64, B=2, S=2048, d_model=1024.
// ---------------------------------------------------------------------------

typedef __attribute__((ext_vector_type(8)))  __bf16 v8bf;
typedef __attribute__((ext_vector_type(16))) __bf16 v16bf;
typedef __attribute__((ext_vector_type(8)))  float  v8f;

typedef __attribute__((ext_vector_type(4))) unsigned int u32x4;
typedef __attribute__((ext_vector_type(8))) int          i32x8;
typedef __attribute__((ext_vector_type(4))) int          i32x4;

#define BATCH 2
#define SEQ   2048
#define HEADS 16
#define DKD   64
#define DM    1024
#define ROWS  (BATCH * SEQ)   /* 4096 */

#if defined(__has_builtin)
#if __has_builtin(__builtin_amdgcn_tensor_load_to_lds) && \
    __has_builtin(__builtin_amdgcn_s_wait_tensorcnt)
#define HAVE_TDM 1
#endif
#if __has_builtin(__builtin_amdgcn_s_cluster_barrier)
#define HAVE_CLUSTER_BARRIER 1
#endif
#endif

static __device__ __forceinline__ v8bf ld_v8(const __bf16* p) {
    return *(const v8bf*)p;              // 16B contiguous
}
static __device__ __forceinline__ v16bf ld_v16(const __bf16* p) {
    return *(const v16bf*)p;             // 32B contiguous
}
static __device__ __forceinline__ v16bf combine16(v8bf lo, v8bf hi) {
    v16bf r;
#pragma unroll
    for (int i = 0; i < 8; ++i) { r[i] = lo[i]; r[i + 8] = hi[i]; }
    return r;
}
static __device__ __forceinline__ v8f wmma_bf16(v16bf a, v16bf b, v8f c) {
    // D = A(16x32 bf16) * B(32x16 bf16) + C(16x16 f32)
    return __builtin_amdgcn_wmma_f32_16x16x32_bf16(
        /*neg_a=*/false, a, /*neg_b=*/false, b,
        /*c_mod=*/(short)0, c, /*reuse_a=*/false, /*reuse_b=*/false);
}

// NULL-tensor TDM op: D# group0.count==0 is defined as a NULL tensor (no-op),
// group0.type=2 ("image") per ISA 8.3. Exercises TENSORcnt path harmlessly.
static __device__ __forceinline__ void tdm_null_op() {
#ifdef HAVE_TDM
    u32x4 g0 = {0u, 0u, 0u, 0x80000000u};   // count=0, type=2
    i32x8 g1 = {0, 0, 0, 0, 0, 0, 0, 0};
    i32x4 g2 = {0, 0, 0, 0};
    i32x4 g3 = {0, 0, 0, 0};
#if __clang_major__ >= 23
    i32x8 g4 = {0, 0, 0, 0, 0, 0, 0, 0};
    __builtin_amdgcn_tensor_load_to_lds(g0, g1, g2, g3, g4, 0);
#else
    __builtin_amdgcn_tensor_load_to_lds(g0, g1, g2, g3, 0);
#endif
    __builtin_amdgcn_s_wait_tensorcnt(0);
#endif
}

// ---------------------------------------------------------------------------
// 1) fp32 -> bf16 conversion
// ---------------------------------------------------------------------------
__global__ void cvt_f32_bf16(const float* __restrict__ in,
                             uint16_t* __restrict__ out_, int n) {
    __bf16* out = (__bf16*)out_;
    int i = blockIdx.x * blockDim.x + threadIdx.x;
    if (i < n) out[i] = (__bf16)in[i];
}

// ---------------------------------------------------------------------------
// 2) GEMM: C[M,N] = A[M,K] * B[N,K]^T   (A, B bf16; C bf16 or f32)
//    block = 128 threads (4 waves); block tile 64(M) x 64(N); K step 32.
//    Register double-buffered: next K-step fragments are loaded before the
//    current WMMAs issue, so s_wait_loadcnt covers a full iteration of slack.
//    Fragment layouts (this code's contiguous k order, shared by A and B):
//      A 16x32 : lane m = lane&15, chunks K in [kA,kA+8) and [kA+8,kA+16),
//                kA = (lane<16) ? 0 : 16   -> two contiguous 16B loads
//      B 32x16 : lane n = lane&15, K in [kB, kB+16), kB = (lane<16) ? 0 : 16
//                -> one contiguous 32B load
// ---------------------------------------------------------------------------
__global__ __launch_bounds__(128)
void gemm_wmma(const uint16_t* __restrict__ A_, const uint16_t* __restrict__ B_,
               float* __restrict__ Cf, uint16_t* __restrict__ Cb_,
               int M, int N, int K, int out_f32) {
    const __bf16* A  = (const __bf16*)A_;
    const __bf16* Bw = (const __bf16*)B_;
    __bf16* Cb = (__bf16*)Cb_;

    const int lane = threadIdx.x & 31;
    const int wave = threadIdx.x >> 5;
    const int lm   = lane & 15;
    const int half = lane >> 4;
    const int kA   = half * 16;
    const int kB   = half * 16;

    const int row0 = blockIdx.y * 64 + wave * 16;
    const int col0 = blockIdx.x * 64;

    v8f acc[4];
#pragma unroll
    for (int t = 0; t < 4; ++t)
#pragma unroll
        for (int v = 0; v < 8; ++v) acc[t][v] = 0.0f;

    const __bf16* arow = A + (size_t)(row0 + lm) * K + kA;
    const __bf16* br0  = Bw + (size_t)(col0 + lm) * K + kB;
    const __bf16* br1  = br0 + (size_t)16 * K;
    const __bf16* br2  = br1 + (size_t)16 * K;
    const __bf16* br3  = br2 + (size_t)16 * K;

    // prologue: fragments for kk = 0
    v16bf af = combine16(ld_v8(arow), ld_v8(arow + 8));
    v16bf b0 = ld_v16(br0), b1 = ld_v16(br1), b2 = ld_v16(br2), b3 = ld_v16(br3);

    for (int kk = 0; kk < K; kk += 32) {
        const int kn = (kk + 32 < K) ? kk + 32 : kk;   // tail reloads (discarded)
        v16bf afn = combine16(ld_v8(arow + kn), ld_v8(arow + kn + 8));
        v16bf b0n = ld_v16(br0 + kn);
        v16bf b1n = ld_v16(br1 + kn);
        v16bf b2n = ld_v16(br2 + kn);
        v16bf b3n = ld_v16(br3 + kn);
        __builtin_prefetch(arow + kk + 256, 0, 1);     // global_prefetch_b8

        acc[0] = wmma_bf16(af, b0, acc[0]);
        acc[1] = wmma_bf16(af, b1, acc[1]);
        acc[2] = wmma_bf16(af, b2, acc[2]);
        acc[3] = wmma_bf16(af, b3, acc[3]);

        af = afn; b0 = b0n; b1 = b1n; b2 = b2n; b3 = b3n;
    }

    // C layout: element (m,n): n = lane&15, m = v + 8*(lane>>4)
#pragma unroll
    for (int t = 0; t < 4; ++t) {
#pragma unroll
        for (int v = 0; v < 8; ++v) {
            const int r = row0 + v + half * 8;
            const int c = col0 + t * 16 + lm;
            if (out_f32) Cf[(size_t)r * N + c] = acc[t][v];
            else         Cb[(size_t)r * N + c] = (__bf16)acc[t][v];
        }
    }
}

// ---------------------------------------------------------------------------
// 3) RoPE (interleaved pairs), optional scale folded in (1/sqrt(DK) for Q).
// ---------------------------------------------------------------------------
__global__ void rope_kernel(uint16_t* __restrict__ buf_, float scale) {
    __bf16* buf = (__bf16*)buf_;
    int gid = blockIdx.x * blockDim.x + threadIdx.x;
    if (gid >= ROWS * (DM / 2)) return;
    const int row = gid >> 9;          // / 512 pairs per row
    const int j   = gid & 511;         // pair index within row
    const int p   = j & 31;            // pair index within head (DK/2 = 32)
    const int s   = row & (SEQ - 1);
    // inv_freq = 10000^(-p/32) = exp(-p * ln(1e4)/32)
    const float inv = __expf(-(float)p * (9.210340371976184f / 32.0f));
    const float ang = (float)s * inv;
    float sn, cs;
    sincosf(ang, &sn, &cs);
    const size_t base = (size_t)row * DM + (size_t)j * 2;
    const float a = (float)buf[base];
    const float b = (float)buf[base + 1];
    buf[base]     = (__bf16)((a * cs - b * sn) * scale);
    buf[base + 1] = (__bf16)((a * sn + b * cs) * scale);
}

// ---------------------------------------------------------------------------
// 4) V transpose: [b*SEQ+s][h*64+dk] -> vt[(h*2+b)*64+dk][s]
// ---------------------------------------------------------------------------
__global__ void transpose_v(const uint16_t* __restrict__ vb_,
                            uint16_t* __restrict__ vt_) {
    const __bf16* vb = (const __bf16*)vb_;
    __bf16* vt = (__bf16*)vt_;
    int gid = blockIdx.x * blockDim.x + threadIdx.x;
    if (gid >= ROWS * DM) return;
    const int row = gid >> 10;
    const int col = gid & 1023;
    const int h  = col >> 6;
    const int dk = col & 63;
    const int b  = row >> 11;
    const int s  = row & (SEQ - 1);
    vt[((size_t)((h * 2 + b) * 64 + dk)) * SEQ + s] = vb[gid];
}

// ---------------------------------------------------------------------------
// 5) Flash attention (causal, online softmax).
//    grid = (SEQ/64, HEADS*BATCH), block = 128 (4 waves).
//    Each wave: 16 queries x full DK=64 accumulator (4 C tiles).
//    Key loop in 32-key tiles. K fragments are double-buffered across tiles
//    and V fragments load immediately after the score WMMAs issue, so global
//    latency overlaps the softmax VALU work and the LDS P-transpose.
// ---------------------------------------------------------------------------
__global__ __launch_bounds__(128)
void flash_attn(const uint16_t* __restrict__ Q_, const uint16_t* __restrict__ K_,
                const uint16_t* __restrict__ Vt_, uint16_t* __restrict__ O_) {
    const __bf16* Q  = (const __bf16*)Q_;
    const __bf16* Kb = (const __bf16*)K_;
    const __bf16* Vt = (const __bf16*)Vt_;
    __bf16* O = (__bf16*)O_;

    const int lane = threadIdx.x & 31;
    const int wave = threadIdx.x >> 5;
    const int lm   = lane & 15;
    const int half = lane >> 4;
    const int kA   = half * 16;    // A-frag K-chunk base
    const int kB   = half * 16;    // B-frag K-chunk base

    const int hb = blockIdx.y;
    const int h  = hb >> 1;
    const int b  = hb & 1;
    const int qbase = blockIdx.x * 64 + wave * 16;

    __shared__ __bf16 psh[4][16][32];   // per-wave P tile (16q x 32k)

    tdm_null_op();                       // NULL-tensor TDM + s_wait_tensorcnt
#ifdef HAVE_CLUSTER_BARRIER
    __builtin_amdgcn_s_cluster_barrier();  // S_NOP when not dispatched in cluster
#endif

    // Q A-fragments (16 x 64 over two K=32 frags), already scaled by 1/8
    const __bf16* qrow = Q + (size_t)(b * SEQ + qbase + lm) * DM + h * 64;
    v16bf qf0 = combine16(ld_v8(qrow + kA),      ld_v8(qrow + kA + 8));
    v16bf qf1 = combine16(ld_v8(qrow + 32 + kA), ld_v8(qrow + 32 + kA + 8));

    // K fragment base: + key*DM selects the key row, + f*32 the dk chunk pair
    const __bf16* kptr = Kb + (size_t)(b * SEQ + lm) * DM + h * 64 + kB;
    // V fragment base: + (tt*16)*SEQ selects dk column group, + key offset
    const __bf16* vptr = Vt + (size_t)((h * 2 + b) * 64 + lm) * SEQ + kB;

    v8f o[4];
#pragma unroll
    for (int t = 0; t < 4; ++t)
#pragma unroll
        for (int v = 0; v < 8; ++v) o[t][v] = 0.0f;
    float rm[8], rs[8];
#pragma unroll
    for (int v = 0; v < 8; ++v) { rm[v] = -3.0e38f; rs[v] = 0.0f; }

    const int ntiles = (qbase + 47) >> 5;   // 32-key tiles covering keys <= qbase+15

    // prologue: K fragments for tile 0
    v16bf kL0 = ld_v16(kptr);
    v16bf kL1 = ld_v16(kptr + 32);
    v16bf kR0 = ld_v16(kptr + (size_t)16 * DM);
    v16bf kR1 = ld_v16(kptr + (size_t)16 * DM + 32);

    for (int t = 0; t < ntiles; ++t) {
        const int kb0 = t * 32;

        // ---- scores: two 16x16 C tiles, K=64 contraction (2 WMMAs each)
        v8f sL, sR;
#pragma unroll
        for (int v = 0; v < 8; ++v) { sL[v] = 0.0f; sR[v] = 0.0f; }
        sL = wmma_bf16(qf0, kL0, sL);
        sL = wmma_bf16(qf1, kL1, sL);
        sR = wmma_bf16(qf0, kR0, sR);
        sR = wmma_bf16(qf1, kR1, sR);

        // ---- issue next tile's K loads + this tile's V loads (overlap softmax)
        const int kn = (t + 1 < ntiles) ? kb0 + 32 : kb0;
        kL0 = ld_v16(kptr + (size_t)kn * DM);
        kL1 = ld_v16(kptr + (size_t)kn * DM + 32);
        kR0 = ld_v16(kptr + (size_t)(kn + 16) * DM);
        kR1 = ld_v16(kptr + (size_t)(kn + 16) * DM + 32);
        v16bf vf[4];
#pragma unroll
        for (int tt = 0; tt < 4; ++tt)
            vf[tt] = ld_v16(vptr + (size_t)(tt * 16) * SEQ + kb0);

        // ---- causal mask in C layout: n = lane&15 (+16 for right tile)
#pragma unroll
        for (int v = 0; v < 8; ++v) {
            const int qi = qbase + v + half * 8;
            if (kb0 + lm > qi)      sL[v] = -3.0e38f;
            if (kb0 + 16 + lm > qi) sR[v] = -3.0e38f;
        }

        // ---- online softmax per row (row == fixed v within a 16-lane half)
#pragma unroll
        for (int v = 0; v < 8; ++v) {
            float tmax = fmaxf(sL[v], sR[v]);
#pragma unroll
            for (int off = 1; off < 16; off <<= 1)
                tmax = fmaxf(tmax, __shfl_xor(tmax, off, 32));
            const float nm = fmaxf(rm[v], tmax);
            const float sc = __expf(rm[v] - nm);
            rm[v] = nm;
            const float pl = __expf(sL[v] - nm);
            const float pr = __expf(sR[v] - nm);
            float ts = pl + pr;
#pragma unroll
            for (int off = 1; off < 16; off <<= 1)
                ts += __shfl_xor(ts, off, 32);
            rs[v] = rs[v] * sc + ts;
#pragma unroll
            for (int tt = 0; tt < 4; ++tt) o[tt][v] *= sc;
            psh[wave][v + half * 8][lm]      = (__bf16)pl;
            psh[wave][v + half * 8][lm + 16] = (__bf16)pr;
        }

        // per-wave LDS ordering: all DS stores done before the gathers below
        asm volatile("s_wait_dscnt 0" ::: "memory");

        // ---- P as A-fragment (16 x 32) from LDS
        const __bf16* prow = &psh[wave][lm][0];
        v16bf pf = combine16(*(const v8bf*)(prow + kA),
                             *(const v8bf*)(prow + kA + 8));

        // ---- PV: 4 B-fragments (32 keys x 16 dk) from transposed V
#pragma unroll
        for (int tt = 0; tt < 4; ++tt)
            o[tt] = wmma_bf16(pf, vf[tt], o[tt]);
    }

    // ---- normalize and store attention output (bf16, [row][h*64+dk])
#pragma unroll
    for (int v = 0; v < 8; ++v) rs[v] = 1.0f / rs[v];
#pragma unroll
    for (int tt = 0; tt < 4; ++tt) {
#pragma unroll
        for (int v = 0; v < 8; ++v) {
            const int r = b * SEQ + qbase + v + half * 8;
            const int c = h * 64 + tt * 16 + lm;
            O[(size_t)r * DM + c] = (__bf16)(o[tt][v] * rs[v]);
        }
    }
}

// ---------------------------------------------------------------------------
// Launch
// ---------------------------------------------------------------------------
extern "C" void kernel_launch(void* const* d_in, const int* in_sizes, int n_in,
                              void* d_out, int out_size, void* d_ws, size_t ws_size,
                              hipStream_t stream) {
    const float* x  = (const float*)d_in[0];
    const float* wq = (const float*)d_in[1];
    const float* wk = (const float*)d_in[2];
    const float* wv = (const float*)d_in[3];
    const float* wo = (const float*)d_in[4];
    float* out = (float*)d_out;

    char* p = (char*)d_ws;
    auto take = [&](size_t bytes) { char* r = p; p += (bytes + 255) & ~(size_t)255; return r; };
    uint16_t* xb  = (uint16_t*)take((size_t)ROWS * DM * 2);
    uint16_t* wqb = (uint16_t*)take((size_t)DM * DM * 2);
    uint16_t* wkb = (uint16_t*)take((size_t)DM * DM * 2);
    uint16_t* wvb = (uint16_t*)take((size_t)DM * DM * 2);
    uint16_t* wob = (uint16_t*)take((size_t)DM * DM * 2);
    uint16_t* qb  = (uint16_t*)take((size_t)ROWS * DM * 2);
    uint16_t* kb  = (uint16_t*)take((size_t)ROWS * DM * 2);
    uint16_t* vb  = (uint16_t*)take((size_t)ROWS * DM * 2);
    uint16_t* vt  = (uint16_t*)take((size_t)ROWS * DM * 2);
    uint16_t* ab  = (uint16_t*)take((size_t)ROWS * DM * 2);

    const int nx = ROWS * DM, nw = DM * DM;
    cvt_f32_bf16<<<(nx + 255) / 256, 256, 0, stream>>>(x,  xb,  nx);
    cvt_f32_bf16<<<(nw + 255) / 256, 256, 0, stream>>>(wq, wqb, nw);
    cvt_f32_bf16<<<(nw + 255) / 256, 256, 0, stream>>>(wk, wkb, nw);
    cvt_f32_bf16<<<(nw + 255) / 256, 256, 0, stream>>>(wv, wvb, nw);
    cvt_f32_bf16<<<(nw + 255) / 256, 256, 0, stream>>>(wo, wob, nw);

    const dim3 ggrid(DM / 64, ROWS / 64);   // (16, 64)
    gemm_wmma<<<ggrid, 128, 0, stream>>>(xb, wqb, nullptr, qb, ROWS, DM, DM, 0);
    gemm_wmma<<<ggrid, 128, 0, stream>>>(xb, wkb, nullptr, kb, ROWS, DM, DM, 0);
    gemm_wmma<<<ggrid, 128, 0, stream>>>(xb, wvb, nullptr, vb, ROWS, DM, DM, 0);

    const int npairs = ROWS * (DM / 2);
    rope_kernel<<<(npairs + 255) / 256, 256, 0, stream>>>(qb, 0.125f); // 1/sqrt(64)
    rope_kernel<<<(npairs + 255) / 256, 256, 0, stream>>>(kb, 1.0f);

    transpose_v<<<(nx + 255) / 256, 256, 0, stream>>>(vb, vt);

    flash_attn<<<dim3(SEQ / 64, HEADS * BATCH), 128, 0, stream>>>(qb, kb, vt, ab);

    gemm_wmma<<<ggrid, 128, 0, stream>>>(ab, wob, out, nullptr, ROWS, DM, DM, 1);
}
